// DETR_91018946937650
// MI455X (gfx1250) — compile-verified, matrix operand order checked by reading, not run on hardware
//
#include <hip/hip_runtime.h>

typedef __attribute__((ext_vector_type(16))) _Float16 v16h;
typedef __attribute__((ext_vector_type(8)))  _Float16 v8h;
typedef __attribute__((ext_vector_type(8)))  float    v8f;

#define B_     16
#define HW_    400
#define D_     256
#define NH_    8
#define DH_    32
#define DFF_   2048
#define CB_    2048
#define NEL_   6
#define NDL_   6
#define KVERB_ 2
#define MAXR_  6
#define NCLS_  9929
#define NVERBS_ 504
#define SCALE_ 0.17677669529663687f  // 1/sqrt(32)

// ---------------------------------------------------------------------------
// WMMA helpers.
// Per-lane fragment layout for 16-bit A (16x32) and B (32x16, stored as N-major
// rows of K): lane L holds row/col (L&15); its 16 halves cover
// K = khalf*8 + [0..7]  (elements 0..7)  and  K = 16 + khalf*8 + [0..7]
// (elements 8..15), khalf = L>>4.  Both 8-half groups are contiguous in K, so
// each fragment is two 16-byte loads.
// ---------------------------------------------------------------------------
__device__ __forceinline__ v16h load_frag(const _Float16* p, int khalf) {
  v8h lo = *(const v8h*)(p + khalf * 8);
  v8h hi = *(const v8h*)(p + 16 + khalf * 8);
  v16h r;
#pragma unroll
  for (int i = 0; i < 8; ++i) { r[i] = lo[i]; r[i + 8] = hi[i]; }
  return r;
}

__device__ __forceinline__ v8f wmma_f16(v16h a, v16h b, v8f c) {
  return __builtin_amdgcn_wmma_f32_16x16x32_f16(false, a, false, b, (short)0, c,
                                                false, false);
}

__device__ __forceinline__ int wmma_kpat(int khalf, int e) {
  return (e < 8) ? (khalf * 8 + e) : (16 + khalf * 8 + (e - 8));
}

// ---------------------------------------------------------------------------
// Generic GEMM:  Y[R,N] = X[R,Kd] @ W[N,Kd]^T + bias  (f16 in, f32 accum)
// One wave per 16x64 output tile (4 accumulators reuse the A fragment).
// Manually 2x-unrolled ping-pong pipeline (Kd is always a multiple of 64 in
// this model): compute with fragment set 0 while set 1 loads, and vice versa.
// No loop-carried buffer rotation -> no register-copy block at the loop tail;
// WMMAs overlap the in-flight b128 loads via partial s_wait_loadcnt.
// ---------------------------------------------------------------------------
__global__ __launch_bounds__(32) void detr_gemm(
    const _Float16* __restrict__ X, const _Float16* __restrict__ W,
    const float* __restrict__ bias, float* __restrict__ Yf,
    _Float16* __restrict__ Yh, int R, int Kd, int N, int relu) {
  const int lane = threadIdx.x;
  const int l15 = lane & 15;
  const int khalf = lane >> 4;
  const int rt = blockIdx.y;
  const int nb = blockIdx.x;

  int row = rt * 16 + l15;
  int arow = row < R ? row : R - 1;
  const _Float16* xp = X + (size_t)arow * Kd;

  const _Float16* wp[4];
#pragma unroll
  for (int t = 0; t < 4; ++t) {
    int col = nb * 64 + t * 16 + l15;
    int bcol = col < N ? col : N - 1;
    wp[t] = W + (size_t)bcol * Kd;
  }

  v8f zero = {};
  v8f acc[4] = {zero, zero, zero, zero};

  // prologue: fragment set 0 for k = 0
  v16h a0 = load_frag(xp, khalf);
  v16h b0[4];
#pragma unroll
  for (int t = 0; t < 4; ++t) b0[t] = load_frag(wp[t], khalf);

  int k = 0;
  while (true) {
    // load set 1 (k+32) while set 0 computes
    v16h a1 = load_frag(xp + k + 32, khalf);
    v16h b1[4];
#pragma unroll
    for (int t = 0; t < 4; ++t) b1[t] = load_frag(wp[t] + k + 32, khalf);
#pragma unroll
    for (int t = 0; t < 4; ++t) acc[t] = wmma_f16(a0, b0[t], acc[t]);

    k += 64;
    const bool more = k < Kd;
    if (more) {
      // reload set 0 (k) while set 1 computes
      a0 = load_frag(xp + k, khalf);
#pragma unroll
      for (int t = 0; t < 4; ++t) b0[t] = load_frag(wp[t] + k, khalf);
    }
#pragma unroll
    for (int t = 0; t < 4; ++t) acc[t] = wmma_f16(a1, b1[t], acc[t]);
    if (!more) break;
  }

#pragma unroll
  for (int t = 0; t < 4; ++t) {
    int cn = nb * 64 + t * 16 + l15;
    if (cn >= N) continue;
    float bv = bias ? bias[cn] : 0.0f;
#pragma unroll
    for (int r = 0; r < 8; ++r) {
      int cm = rt * 16 + r + khalf * 8;
      if (cm >= R) continue;
      float v = acc[t][r] + bv;
      if (relu) v = fmaxf(v, 0.0f);
      size_t off = (size_t)cm * N + cn;
      if (Yf) Yf[off] = v;
      if (Yh) Yh[off] = (_Float16)v;
    }
  }
}

// ---------------------------------------------------------------------------
// Encoder self-attention.  One wave per (qtile, head, batch).
// S = Q K^T * scale + bias : dh=32 == WMMA K -> one WMMA per 16x16 score tile.
// ---------------------------------------------------------------------------
__global__ __launch_bounds__(32) void detr_attn_enc(
    const _Float16* __restrict__ QH, const _Float16* __restrict__ KH,
    const _Float16* __restrict__ VH, const unsigned char* __restrict__ mask,
    _Float16* __restrict__ OH) {
  __shared__ float sc[16 * 400];
  __shared__ _Float16 pr[16 * 416];
  const int lane = threadIdx.x;
  const int l15 = lane & 15;
  const int khalf = lane >> 4;
  const int qt = blockIdx.x, h = blockIdx.y, b = blockIdx.z;

  const _Float16* qbase =
      QH + (size_t)(b * HW_ + qt * 16 + l15) * D_ + h * DH_;
  v16h aq = load_frag(qbase, khalf);

  for (int nt = 0; nt < 25; ++nt) {
    const _Float16* kbp =
        KH + (size_t)(b * HW_ + nt * 16 + l15) * D_ + h * DH_;
    v16h bk = load_frag(kbp, khalf);
    v8f c = {};
    c = wmma_f16(aq, bk, c);
    int key = nt * 16 + l15;
    float kb = mask[b * HW_ + key] ? -1e9f : 0.0f;
#pragma unroll
    for (int r = 0; r < 8; ++r) sc[(r + khalf * 8) * 400 + key] = c[r] * SCALE_ + kb;
  }
  __syncthreads();

  if (lane < 16) {
    float mx = -1e30f;
    for (int j = 0; j < 400; ++j) mx = fmaxf(mx, sc[lane * 400 + j]);
    float s = 0.0f;
    for (int j = 0; j < 400; ++j) {
      float e = __expf(sc[lane * 400 + j] - mx);
      sc[lane * 400 + j] = e;
      s += e;
    }
    float inv = 1.0f / s;
    for (int j = 0; j < 400; ++j) pr[lane * 416 + j] = (_Float16)(sc[lane * 400 + j] * inv);
    for (int j = 400; j < 416; ++j) pr[lane * 416 + j] = (_Float16)0.0f;
  }
  __syncthreads();

  // O = P @ V   (K loop over 13 tiles of 32 keys; dh=32 -> two 16-col tiles)
#pragma unroll
  for (int ct = 0; ct < 2; ++ct) {
    v8f o = {};
    for (int kt = 0; kt < 13; ++kt) {
      v16h ap = load_frag(&pr[l15 * 416 + kt * 32], khalf);
      v16h bv;
#pragma unroll
      for (int e = 0; e < 16; ++e) {
        int key = kt * 32 + wmma_kpat(khalf, e);
        bv[e] = (key < 400)
                    ? VH[(size_t)(b * HW_ + key) * D_ + h * DH_ + ct * 16 + l15]
                    : (_Float16)0.0f;
      }
      o = wmma_f16(ap, bv, o);
    }
#pragma unroll
    for (int r = 0; r < 8; ++r) {
      int m = r + khalf * 8;
      OH[(size_t)(b * HW_ + qt * 16 + m) * D_ + h * DH_ + ct * 16 + l15] =
          (_Float16)o[r];
    }
  }
}

// ---------------------------------------------------------------------------
// Decoder cross-attention: 6 queries (padded to a 16-row tile) vs 400 keys.
// ---------------------------------------------------------------------------
__global__ __launch_bounds__(32) void detr_attn_cross(
    const _Float16* __restrict__ QH, const _Float16* __restrict__ KH,
    const _Float16* __restrict__ VH, const unsigned char* __restrict__ mask,
    _Float16* __restrict__ OH) {
  __shared__ float sc[16 * 400];
  __shared__ _Float16 pr[16 * 416];
  const int lane = threadIdx.x;
  const int l15 = lane & 15;
  const int khalf = lane >> 4;
  const int h = blockIdx.x, b = blockIdx.y;

  int qrow = l15 < MAXR_ ? l15 : MAXR_ - 1;
  const _Float16* qbase = QH + (size_t)(b * MAXR_ + qrow) * D_ + h * DH_;
  v16h aq = load_frag(qbase, khalf);

  for (int nt = 0; nt < 25; ++nt) {
    const _Float16* kbp =
        KH + (size_t)(b * HW_ + nt * 16 + l15) * D_ + h * DH_;
    v16h bk = load_frag(kbp, khalf);
    v8f c = {};
    c = wmma_f16(aq, bk, c);
    int key = nt * 16 + l15;
    float kb = mask[b * HW_ + key] ? -1e9f : 0.0f;
#pragma unroll
    for (int r = 0; r < 8; ++r) sc[(r + khalf * 8) * 400 + key] = c[r] * SCALE_ + kb;
  }
  __syncthreads();

  if (lane < 16) {
    float mx = -1e30f;
    for (int j = 0; j < 400; ++j) mx = fmaxf(mx, sc[lane * 400 + j]);
    float s = 0.0f;
    for (int j = 0; j < 400; ++j) {
      float e = __expf(sc[lane * 400 + j] - mx);
      sc[lane * 400 + j] = e;
      s += e;
    }
    float inv = 1.0f / s;
    for (int j = 0; j < 400; ++j) pr[lane * 416 + j] = (_Float16)(sc[lane * 400 + j] * inv);
    for (int j = 400; j < 416; ++j) pr[lane * 416 + j] = (_Float16)0.0f;
  }
  __syncthreads();

#pragma unroll
  for (int ct = 0; ct < 2; ++ct) {
    v8f o = {};
    for (int kt = 0; kt < 13; ++kt) {
      v16h ap = load_frag(&pr[l15 * 416 + kt * 32], khalf);
      v16h bv;
#pragma unroll
      for (int e = 0; e < 16; ++e) {
        int key = kt * 32 + wmma_kpat(khalf, e);
        bv[e] = (key < 400)
                    ? VH[(size_t)(b * HW_ + key) * D_ + h * DH_ + ct * 16 + l15]
                    : (_Float16)0.0f;
      }
      o = wmma_f16(ap, bv, o);
    }
#pragma unroll
    for (int r = 0; r < 8; ++r) {
      int m = r + khalf * 8;
      if (m < MAXR_)
        OH[(size_t)(b * MAXR_ + m) * D_ + h * DH_ + ct * 16 + l15] =
            (_Float16)o[r];
    }
  }
}

// ---------------------------------------------------------------------------
// Decoder self-attention (6x6, masked by role validity) — tiny, plain VALU.
// ---------------------------------------------------------------------------
__global__ void detr_attn_dec_self(const _Float16* __restrict__ Q,
                                   const _Float16* __restrict__ Kt,
                                   const _Float16* __restrict__ V,
                                   const int* __restrict__ verbs,
                                   const int* __restrict__ lens, int kk,
                                   _Float16* __restrict__ O) {
  int idx = blockIdx.x * blockDim.x + threadIdx.x;
  if (idx >= B_ * NH_ * MAXR_) return;
  int q = idx % MAXR_;
  int h = (idx / MAXR_) % NH_;
  int b = idx / (MAXR_ * NH_);
  int vb = verbs[b * KVERB_ + kk];
  int len = lens[vb];

  float sc[MAXR_];
  float mx = -1e30f;
#pragma unroll
  for (int j = 0; j < MAXR_; ++j) {
    float s = 0.0f;
    for (int d = 0; d < DH_; ++d)
      s += (float)Q[(size_t)(b * MAXR_ + q) * D_ + h * DH_ + d] *
           (float)Kt[(size_t)(b * MAXR_ + j) * D_ + h * DH_ + d];
    s *= SCALE_;
    if (j >= len) s = -1e9f;
    sc[j] = s;
    mx = fmaxf(mx, s);
  }
  float sum = 0.0f;
#pragma unroll
  for (int j = 0; j < MAXR_; ++j) { sc[j] = __expf(sc[j] - mx); sum += sc[j]; }
  float inv = 1.0f / sum;
  for (int d = 0; d < DH_; ++d) {
    float o = 0.0f;
#pragma unroll
    for (int j = 0; j < MAXR_; ++j)
      o += sc[j] * (float)V[(size_t)(b * MAXR_ + j) * D_ + h * DH_ + d];
    O[(size_t)(b * MAXR_ + q) * D_ + h * DH_ + d] = (_Float16)(o * inv);
  }
}

// ---------------------------------------------------------------------------
// Residual + LayerNorm over D=256 (one block per row), optional row zero-mask.
// ---------------------------------------------------------------------------
__global__ void detr_resid_ln(const float* __restrict__ x,
                              const float* __restrict__ add,
                              const float* __restrict__ s,
                              const float* __restrict__ bb,
                              float* __restrict__ outF,
                              _Float16* __restrict__ outH,
                              const unsigned char* __restrict__ rowmask) {
  int row = blockIdx.x;
  int t = threadIdx.x;  // 256 threads == D
  size_t off = (size_t)row * D_ + t;
  float v = x[off] + (add ? add[off] : 0.0f);
  __shared__ float red[D_];
  red[t] = v;
  __syncthreads();
  for (int s2 = 128; s2 > 0; s2 >>= 1) {
    if (t < s2) red[t] += red[t + s2];
    __syncthreads();
  }
  float mean = red[0] * (1.0f / D_);
  __syncthreads();
  float d = v - mean;
  red[t] = d * d;
  __syncthreads();
  for (int s2 = 128; s2 > 0; s2 >>= 1) {
    if (t < s2) red[t] += red[t + s2];
    __syncthreads();
  }
  float var = red[0] * (1.0f / D_);
  float y = d * rsqrtf(var + 1e-5f) * s[t] + bb[t];
  if (rowmask && !rowmask[row]) y = 0.0f;
  if (outF) outF[off] = y;
  if (outH) outH[off] = (_Float16)y;
}

// ---------------------------------------------------------------------------
// Small utility kernels.
// ---------------------------------------------------------------------------
__global__ void detr_cvt(const float* __restrict__ in, _Float16* __restrict__ o,
                         long n) {
  long i = blockIdx.x * (long)blockDim.x + threadIdx.x;
  long st = gridDim.x * (long)blockDim.x;
  for (; i < n; i += st) o[i] = (_Float16)in[i];
}

__global__ void detr_add_cvt(const float* __restrict__ a,
                             const float* __restrict__ b,
                             _Float16* __restrict__ oh, long n) {
  long i = blockIdx.x * (long)blockDim.x + threadIdx.x;
  long st = gridDim.x * (long)blockDim.x;
  for (; i < n; i += st) oh[i] = (_Float16)(a[i] + (b ? b[i] : 0.0f));
}

__global__ void detr_fill0(float* __restrict__ f, _Float16* __restrict__ h,
                           long n) {
  long i = blockIdx.x * (long)blockDim.x + threadIdx.x;
  long st = gridDim.x * (long)blockDim.x;
  for (; i < n; i += st) {
    if (f) f[i] = 0.0f;
    if (h) h[i] = (_Float16)0.0f;
  }
}

// src (B,CB,HW) f32 -> srcT (B,HW,CB) f16 via LDS tile transpose
__global__ void detr_transpose_src(const float* __restrict__ src,
                                   _Float16* __restrict__ dst) {
  __shared__ float tile[32][33];
  int b = blockIdx.z;
  int c0 = blockIdx.y * 32;
  int w0 = blockIdx.x * 32;
  int tx = threadIdx.x, ty = threadIdx.y;  // 32 x 8
  for (int i = ty; i < 32; i += 8) {
    int c = c0 + i, w = w0 + tx;
    tile[i][tx] = (w < HW_) ? src[((size_t)b * CB_ + c) * HW_ + w] : 0.0f;
  }
  __syncthreads();
  for (int i = ty; i < 32; i += 8) {
    int w = w0 + i, c = c0 + tx;
    if (w < HW_)
      dst[((size_t)b * HW_ + w) * CB_ + c] = (_Float16)tile[tx][i];
  }
}

__global__ void detr_build_qpos(const int* __restrict__ verbs,
                                const int* __restrict__ roles,
                                const int* __restrict__ lens,
                                const float* __restrict__ role_w,
                                const float* __restrict__ verb_w, int kk,
                                float* __restrict__ qposF,
                                _Float16* __restrict__ qposH,
                                unsigned char* __restrict__ rowvalid) {
  int idx = blockIdx.x * 256 + threadIdx.x;  // 96*256 total
  int d = idx & 255;
  int r = (idx >> 8) % MAXR_;
  int b = idx / (256 * MAXR_);
  int vb = verbs[b * KVERB_ + kk];
  int role = roles[vb * MAXR_ + r];
  float v = (d < 128) ? role_w[role * 128 + d] : verb_w[vb * 128 + (d - 128)];
  qposF[idx] = v;
  qposH[idx] = (_Float16)v;
  if (d == 0) rowvalid[b * MAXR_ + r] = (r < lens[vb]) ? 1 : 0;
}

__global__ void detr_mean_pool(const float* __restrict__ mem,
                               _Float16* __restrict__ outH) {
  int b = blockIdx.x;
  int d = threadIdx.x;
  float s = 0.0f;
  for (int hw = 0; hw < HW_; ++hw) s += mem[(size_t)(b * HW_ + hw) * D_ + d];
  outH[b * D_ + d] = (_Float16)(s * (1.0f / HW_));
}

// ---------------------------------------------------------------------------
// Host orchestration
// ---------------------------------------------------------------------------
extern "C" void kernel_launch(void* const* d_in, const int* in_sizes, int n_in,
                              void* d_out, int out_size, void* d_ws,
                              size_t ws_size, hipStream_t stream) {
  (void)in_sizes; (void)n_in; (void)out_size; (void)ws_size;

  const float* src = (const float*)d_in[0];
  const float* pos = (const float*)d_in[1];
  const unsigned char* mask = (const unsigned char*)d_in[2];
  const int* verbs = (const int*)d_in[3];
  const int* roles = (const int*)d_in[4];
  const int* lens = (const int*)d_in[5];
  const float* ipw = (const float*)d_in[6];
  const float* ipb = (const float*)d_in[7];
  const float* role_w = (const float*)d_in[8];
  const float* verb_w = (const float*)d_in[9];
  const float* enc_Wqkv = (const float*)d_in[10];
  const float* enc_bqkv = (const float*)d_in[11];
  const float* enc_Wo = (const float*)d_in[12];
  const float* enc_bo = (const float*)d_in[13];
  const float* enc_W1 = (const float*)d_in[14];
  const float* enc_b1 = (const float*)d_in[15];
  const float* enc_W2 = (const float*)d_in[16];
  const float* enc_b2 = (const float*)d_in[17];
  const float* enc_ln_s = (const float*)d_in[18];
  const float* enc_ln_b = (const float*)d_in[19];
  const float* dec_sWqkv = (const float*)d_in[20];
  const float* dec_sbqkv = (const float*)d_in[21];
  const float* dec_sWo = (const float*)d_in[22];
  const float* dec_sbo = (const float*)d_in[23];
  const float* dec_cWqkv = (const float*)d_in[24];
  const float* dec_cbqkv = (const float*)d_in[25];
  const float* dec_cWo = (const float*)d_in[26];
  const float* dec_cbo = (const float*)d_in[27];
  const float* dec_W1 = (const float*)d_in[28];
  const float* dec_b1 = (const float*)d_in[29];
  const float* dec_W2 = (const float*)d_in[30];
  const float* dec_b2 = (const float*)d_in[31];
  const float* dec_ln_s = (const float*)d_in[32];
  const float* dec_ln_b = (const float*)d_in[33];
  const float* fin_s = (const float*)d_in[34];
  const float* fin_b = (const float*)d_in[35];
  const float* cls_w = (const float*)d_in[36];
  const float* cls_b = (const float*)d_in[37];
  const float* vcls_w = (const float*)d_in[38];
  const float* vcls_b = (const float*)d_in[39];
  float* out = (float*)d_out;

  char* cur = (char*)d_ws;
  auto alloc = [&](size_t bytes) -> char* {
    char* p = cur;
    cur += (bytes + 255) & ~(size_t)255;
    return p;
  };

  const size_t NROW = (size_t)B_ * HW_;    // 6400
  const size_t NDEC = (size_t)B_ * MAXR_;  // 96

  // f16 weights
  _Float16* ipwH = (_Float16*)alloc((size_t)D_ * CB_ * 2);
  _Float16* encWqkvH = (_Float16*)alloc((size_t)NEL_ * 3 * D_ * D_ * 2);
  _Float16* encWoH = (_Float16*)alloc((size_t)NEL_ * D_ * D_ * 2);
  _Float16* encW1H = (_Float16*)alloc((size_t)NEL_ * DFF_ * D_ * 2);
  _Float16* encW2H = (_Float16*)alloc((size_t)NEL_ * D_ * DFF_ * 2);
  _Float16* decSWqkvH = (_Float16*)alloc((size_t)NDL_ * 3 * D_ * D_ * 2);
  _Float16* decSWoH = (_Float16*)alloc((size_t)NDL_ * D_ * D_ * 2);
  _Float16* decCWqkvH = (_Float16*)alloc((size_t)NDL_ * 3 * D_ * D_ * 2);
  _Float16* decCWoH = (_Float16*)alloc((size_t)NDL_ * D_ * D_ * 2);
  _Float16* decW1H = (_Float16*)alloc((size_t)NDL_ * DFF_ * D_ * 2);
  _Float16* decW2H = (_Float16*)alloc((size_t)NDL_ * D_ * DFF_ * 2);
  _Float16* clsWH = (_Float16*)alloc((size_t)NCLS_ * D_ * 2);
  _Float16* vclsWH = (_Float16*)alloc((size_t)NVERBS_ * D_ * 2);
  // srcT (B,HW,CB) f16; reused as FFN hidden (6400 x 2048 f16) after input proj
  _Float16* srcTH = (_Float16*)alloc(NROW * CB_ * 2);
  _Float16* hidH = srcTH;
  // activations
  float* memF = (float*)alloc(NROW * D_ * 4);
  _Float16* memH = (_Float16*)alloc(NROW * D_ * 2);
  _Float16* posH = (_Float16*)alloc(NROW * D_ * 2);  // qk (enc) / mem_pos (dec)
  _Float16* QH = (_Float16*)alloc(NROW * D_ * 2);
  _Float16* KH = (_Float16*)alloc(NROW * D_ * 2);
  _Float16* VH = (_Float16*)alloc(NROW * D_ * 2);
  _Float16* attnH = (_Float16*)alloc(NROW * D_ * 2);
  float* projF = (float*)alloc(NROW * D_ * 4);
  // decoder small buffers
  float* qposF = (float*)alloc(NDEC * D_ * 4);
  _Float16* qposH = (_Float16*)alloc(NDEC * D_ * 2);
  float* tgtF = (float*)alloc(NDEC * D_ * 4);
  _Float16* tgtH = (_Float16*)alloc(NDEC * D_ * 2);
  _Float16* qinH = (_Float16*)alloc(NDEC * D_ * 2);
  _Float16* QsH = (_Float16*)alloc(NDEC * D_ * 2);
  _Float16* KsH = (_Float16*)alloc(NDEC * D_ * 2);
  _Float16* VsH = (_Float16*)alloc(NDEC * D_ * 2);
  _Float16* hsH = (_Float16*)alloc(NDEC * D_ * 2);
  _Float16* pooledH = (_Float16*)alloc((size_t)B_ * D_ * 2);
  unsigned char* rowvalid = (unsigned char*)alloc(NDEC);

  auto cvt = [&](const float* in, _Float16* o, long n) {
    int blocks = (int)((n + 255) / 256);
    if (blocks > 8192) blocks = 8192;
    detr_cvt<<<blocks, 256, 0, stream>>>(in, o, n);
  };
  auto addcvt = [&](const float* a, const float* b, _Float16* o, long n) {
    int blocks = (int)((n + 255) / 256);
    if (blocks > 8192) blocks = 8192;
    detr_add_cvt<<<blocks, 256, 0, stream>>>(a, b, o, n);
  };
  auto gemm = [&](const _Float16* X, const _Float16* W, const float* bias,
                  float* Yf, _Float16* Yh, int R, int Kd, int N, int relu) {
    dim3 g((N + 63) / 64, (R + 15) / 16);
    detr_gemm<<<g, 32, 0, stream>>>(X, W, bias, Yf, Yh, R, Kd, N, relu);
  };

  // ---- weight conversion ----
  cvt(ipw, ipwH, (long)D_ * CB_);
  cvt(enc_Wqkv, encWqkvH, (long)NEL_ * 3 * D_ * D_);
  cvt(enc_Wo, encWoH, (long)NEL_ * D_ * D_);
  cvt(enc_W1, encW1H, (long)NEL_ * DFF_ * D_);
  cvt(enc_W2, encW2H, (long)NEL_ * D_ * DFF_);
  cvt(dec_sWqkv, decSWqkvH, (long)NDL_ * 3 * D_ * D_);
  cvt(dec_sWo, decSWoH, (long)NDL_ * D_ * D_);
  cvt(dec_cWqkv, decCWqkvH, (long)NDL_ * 3 * D_ * D_);
  cvt(dec_cWo, decCWoH, (long)NDL_ * D_ * D_);
  cvt(dec_W1, decW1H, (long)NDL_ * DFF_ * D_);
  cvt(dec_W2, decW2H, (long)NDL_ * D_ * DFF_);
  cvt(cls_w, clsWH, (long)NCLS_ * D_);
  cvt(vcls_w, vclsWH, (long)NVERBS_ * D_);

  // ---- input projection: mem = srcT @ ipw^T + ipb ----
  detr_transpose_src<<<dim3((HW_ + 31) / 32, CB_ / 32, B_), dim3(32, 8), 0,
                       stream>>>(src, srcTH);
  gemm(srcTH, ipwH, ipb, memF, memH, (int)NROW, CB_, D_, 0);

  // ---- encoder ----
  for (int i = 0; i < NEL_; ++i) {
    addcvt(memF, pos, posH, (long)NROW * D_);  // qk = mem + pos
    gemm(posH, encWqkvH + (size_t)(i * 3 + 0) * D_ * D_,
         enc_bqkv + (size_t)(i * 3 + 0) * D_, nullptr, QH, (int)NROW, D_, D_, 0);
    gemm(posH, encWqkvH + (size_t)(i * 3 + 1) * D_ * D_,
         enc_bqkv + (size_t)(i * 3 + 1) * D_, nullptr, KH, (int)NROW, D_, D_, 0);
    gemm(memH, encWqkvH + (size_t)(i * 3 + 2) * D_ * D_,
         enc_bqkv + (size_t)(i * 3 + 2) * D_, nullptr, VH, (int)NROW, D_, D_, 0);
    detr_attn_enc<<<dim3(HW_ / 16, NH_, B_), 32, 0, stream>>>(QH, KH, VH, mask,
                                                              attnH);
    gemm(attnH, encWoH + (size_t)i * D_ * D_, enc_bo + (size_t)i * D_, projF,
         nullptr, (int)NROW, D_, D_, 0);
    detr_resid_ln<<<(int)NROW, D_, 0, stream>>>(
        memF, projF, enc_ln_s + (size_t)(i * 2 + 0) * D_,
        enc_ln_b + (size_t)(i * 2 + 0) * D_, memF, memH, nullptr);
    gemm(memH, encW1H + (size_t)i * DFF_ * D_, enc_b1 + (size_t)i * DFF_,
         nullptr, hidH, (int)NROW, D_, DFF_, 1);
    gemm(hidH, encW2H + (size_t)i * D_ * DFF_, enc_b2 + (size_t)i * D_, projF,
         nullptr, (int)NROW, DFF_, D_, 0);
    detr_resid_ln<<<(int)NROW, D_, 0, stream>>>(
        memF, projF, enc_ln_s + (size_t)(i * 2 + 1) * D_,
        enc_ln_b + (size_t)(i * 2 + 1) * D_, memF, memH, nullptr);
  }

  // ---- verb head: mean-pool + classifier ----
  detr_mean_pool<<<B_, D_, 0, stream>>>(memF, pooledH);
  gemm(pooledH, vclsWH, vcls_b, out, nullptr, B_, D_, NVERBS_, 0);

  // mem_pos (fixed for decoder)
  addcvt(memF, pos, posH, (long)NROW * D_);

  // ---- decoder (K=2 verb hypotheses) ----
  for (int kk = 0; kk < KVERB_; ++kk) {
    detr_build_qpos<<<(int)NDEC, 256, 0, stream>>>(verbs, roles, lens, role_w,
                                                   verb_w, kk, qposF, qposH,
                                                   rowvalid);
    detr_fill0<<<96, 256, 0, stream>>>(tgtF, tgtH, (long)NDEC * D_);

    for (int i = 0; i < NDL_; ++i) {
      // --- self attention ---
      addcvt(tgtF, qposF, qinH, (long)NDEC * D_);
      gemm(qinH, decSWqkvH + (size_t)(i * 3 + 0) * D_ * D_,
           dec_sbqkv + (size_t)(i * 3 + 0) * D_, nullptr, QsH, (int)NDEC, D_,
           D_, 0);
      gemm(qinH, decSWqkvH + (size_t)(i * 3 + 1) * D_ * D_,
           dec_sbqkv + (size_t)(i * 3 + 1) * D_, nullptr, KsH, (int)NDEC, D_,
           D_, 0);
      gemm(tgtH, decSWqkvH + (size_t)(i * 3 + 2) * D_ * D_,
           dec_sbqkv + (size_t)(i * 3 + 2) * D_, nullptr, VsH, (int)NDEC, D_,
           D_, 0);
      detr_attn_dec_self<<<3, 256, 0, stream>>>(QsH, KsH, VsH, verbs, lens, kk,
                                                attnH);
      gemm(attnH, decSWoH + (size_t)i * D_ * D_, dec_sbo + (size_t)i * D_,
           projF, nullptr, (int)NDEC, D_, D_, 0);
      detr_resid_ln<<<(int)NDEC, D_, 0, stream>>>(
          tgtF, projF, dec_ln_s + (size_t)(i * 3 + 0) * D_,
          dec_ln_b + (size_t)(i * 3 + 0) * D_, tgtF, tgtH, nullptr);
      // --- cross attention ---
      addcvt(tgtF, qposF, qinH, (long)NDEC * D_);
      gemm(qinH, decCWqkvH + (size_t)(i * 3 + 0) * D_ * D_,
           dec_cbqkv + (size_t)(i * 3 + 0) * D_, nullptr, QsH, (int)NDEC, D_,
           D_, 0);
      gemm(posH, decCWqkvH + (size_t)(i * 3 + 1) * D_ * D_,
           dec_cbqkv + (size_t)(i * 3 + 1) * D_, nullptr, KH, (int)NROW, D_,
           D_, 0);
      gemm(memH, decCWqkvH + (size_t)(i * 3 + 2) * D_ * D_,
           dec_cbqkv + (size_t)(i * 3 + 2) * D_, nullptr, VH, (int)NROW, D_,
           D_, 0);
      detr_attn_cross<<<dim3(NH_, B_), 32, 0, stream>>>(QsH, KH, VH, mask,
                                                        attnH);
      gemm(attnH, decCWoH + (size_t)i * D_ * D_, dec_cbo + (size_t)i * D_,
           projF, nullptr, (int)NDEC, D_, D_, 0);
      detr_resid_ln<<<(int)NDEC, D_, 0, stream>>>(
          tgtF, projF, dec_ln_s + (size_t)(i * 3 + 1) * D_,
          dec_ln_b + (size_t)(i * 3 + 1) * D_, tgtF, tgtH, nullptr);
      // --- FFN ---
      gemm(tgtH, decW1H + (size_t)i * DFF_ * D_, dec_b1 + (size_t)i * DFF_,
           nullptr, hidH, (int)NDEC, D_, DFF_, 1);
      gemm(hidH, decW2H + (size_t)i * D_ * DFF_, dec_b2 + (size_t)i * D_,
           projF, nullptr, (int)NDEC, DFF_, D_, 0);
      detr_resid_ln<<<(int)NDEC, D_, 0, stream>>>(
          tgtF, projF, dec_ln_s + (size_t)(i * 3 + 2) * D_,
          dec_ln_b + (size_t)(i * 3 + 2) * D_, tgtF, tgtH, nullptr);
    }

    // hs = LN(tgt; fin) * valid ; logits = hs @ cls_w^T + cls_b
    detr_resid_ln<<<(int)NDEC, D_, 0, stream>>>(tgtF, nullptr, fin_s, fin_b,
                                                nullptr, hsH, rowvalid);
    gemm(hsH, clsWH, cls_b,
         out + (size_t)B_ * NVERBS_ + (size_t)kk * NDEC * NCLS_, nullptr,
         (int)NDEC, D_, NCLS_, 0);
  }
}
